// CANE_22746146800273
// MI455X (gfx1250) — matrix operand order repaired and checked by reading, not compiled, not run on hardware
//
#include <hip/hip_runtime.h>

typedef __attribute__((ext_vector_type(16))) _Float16 v16h;
typedef __attribute__((ext_vector_type(8)))  _Float16 v8h;
typedef __attribute__((ext_vector_type(8)))  float    v8f;

#define BB   512
#define LL   300
#define LM1  299
#define EE   100
#define CC   100
#define LP   304   // padded L-1 (19 * 16)
#define CP   128   // padded C   (4 * 32 WMMA K-steps)
#define KK   200   // conv K (2*E)
#define KP   224   // padded conv K (7 * 32)
#define RHO1 1.0f
#define RHO2 0.3f
#define RHO3 0.3f

// ---- workspace layout (bytes, all offsets 256-aligned) ----
static constexpr size_t WP_OFF  = 0;                              // f16 [128][224] conv weights, c-major
static constexpr size_t RT_OFF  = (size_t)128 * KP * 2;           // 57344: f16 [128][128] rand^T (d-major)
static constexpr size_t BIAS_OFF= RT_OFF + (size_t)128 * 128 * 2; // 90112: f32 [128] padded bias
static constexpr size_t HA_OFF  = BIAS_OFF + 512;                 // 90624
static constexpr size_t HSZ     = (size_t)BB * LP * CP * 2;       // 39,845,888 per h tensor (f16)
static constexpr size_t HB_OFF  = HA_OFF + HSZ;
static constexpr size_t HN_OFF  = HB_OFF + HSZ;
static constexpr size_t HR_OFF  = HN_OFF + HSZ;
static constexpr size_t RS1_OFF = HR_OFF + HSZ;
static constexpr size_t SSZ     = (size_t)BB * LP * 4;            // f32 [B][LP]
static constexpr size_t CS1_OFF = RS1_OFF + SSZ;
static constexpr size_t CS3_OFF = CS1_OFF + SSZ;

static constexpr unsigned ATT_LDS  = 2u * LP * CP * 2 + 2u * LP * 4;  // 158080 bytes
static constexpr unsigned CONV_LDS = (unsigned)LP * KP * 2;           // 136192 bytes

__device__ __forceinline__ v16h cat16(v8h lo, v8h hi) {
  return __builtin_shufflevector(lo, hi, 0,1,2,3,4,5,6,7,8,9,10,11,12,13,14,15);
}

__device__ __forceinline__ v8f wmma16(v16h a, v16h b, v8f c) {
  // D = A(16x32 f16) * B(32x16 f16) + C(16x16 f32)
  return __builtin_amdgcn_wmma_f32_16x16x32_f16(false, a, false, b, (short)0, c, false, false);
}

// -------------------------------------------------------------------
// K0: repack conv weights / rand matrix / bias into padded, operand-
// friendly layouts (all padding explicitly zeroed).
// -------------------------------------------------------------------
__global__ void __launch_bounds__(256) prep_kernel(
    const float* __restrict__ conv_w, const float* __restrict__ rand_m,
    const float* __restrict__ conv_b,
    _Float16* __restrict__ Wp, _Float16* __restrict__ randT,
    float* __restrict__ biasP)
{
  int idx0 = blockIdx.x * 256 + threadIdx.x;
  int stride = gridDim.x * 256;
  for (int i = idx0; i < 128 * KP; i += stride) {
    int c = i / KP, k = i - c * KP;
    Wp[i] = (c < CC && k < KK) ? (_Float16)conv_w[c * KK + k] : (_Float16)0.f;
  }
  for (int i = idx0; i < 128 * 128; i += stride) {
    int d = i >> 7, c = i & 127;
    randT[i] = (c < CC && d < CC) ? (_Float16)rand_m[c * CC + d] : (_Float16)0.f;
  }
  for (int i = idx0; i < 128; i += stride)
    biasP[i] = (i < CC) ? conv_b[i] : 0.f;
}

__global__ void zero_out_kernel(float* out) {
  if (blockIdx.x == 0 && threadIdx.x == 0) out[0] = 0.f;
}

// -------------------------------------------------------------------
// K1: gather + conv(window 2) + tanh for one text tensor per blockIdx.y.
// Builds the padded im2col matrix X[304][224] f16 directly in dynamic
// LDS (zeros in all padding), so every WMMA A-operand fetch is two
// unconditional 16B ds_load_b128 -- no per-element predication.
// Per batch: GEMM [299 x 200] x [200 x 100] via WMMA f16.
// Output h stored f16 [b][l][c], zero-padded to [304][128].
// -------------------------------------------------------------------
__global__ void __launch_bounds__(256) conv_kernel(
    const int* __restrict__ Ta, const int* __restrict__ Tb, const int* __restrict__ Tn,
    const float* __restrict__ text_emb, const float* __restrict__ biasP,
    const _Float16* __restrict__ Wp,
    _Float16* __restrict__ hA, _Float16* __restrict__ hB, _Float16* __restrict__ hN)
{
  __shared__ int ids[LL];
  extern __shared__ char smem[];
  _Float16* X = (_Float16*)smem;   // [304][224] im2col, padded with zeros

  const int b = blockIdx.x, y = blockIdx.y, tid = threadIdx.x;
  const int* Tx = (y == 0) ? Ta : (y == 1) ? Tb : Tn;
  _Float16* hout = (y == 0) ? hA : (y == 1) ? hB : hN;

  for (int idx = tid; idx < LL; idx += 256) ids[idx] = Tx[b * LL + idx];
  __syncthreads();

  for (int idx = tid; idx < LP * KP; idx += 256) {
    int l = idx / KP, k = idx - l * KP;
    float v = 0.f;
    if (l < LM1 && k < KK) {
      int row = (k < EE) ? l : l + 1;
      int e   = (k < EE) ? k : k - EE;
      v = text_emb[(size_t)ids[row] * EE + e];
    }
    X[idx] = (_Float16)v;
  }
  __syncthreads();

  const int lane = tid & 31, wv = tid >> 5;
  const int col = lane & 15, half = lane >> 4;
  const int kb8 = half * 8, kb16 = half * 16, rbase = half * 8;

  for (int t = wv; t < 19 * 8; t += 8) {          // 19 m-tiles x 8 n-tiles
    const int mt = t >> 3, nt = t & 7;
    v8f acc = {};
    #pragma unroll
    for (int ks = 0; ks < 7; ++ks) {              // K = 224 (padded 200)
      const _Float16* pa = X + (mt * 16 + col) * KP + ks * 32 + kb8;
      v16h am = cat16(*(const v8h*)pa, *(const v8h*)(pa + 16));
      const _Float16* pb = Wp + (size_t)(nt * 16 + col) * KP + ks * 32 + kb8;
      v16h bm = cat16(*(const v8h*)pb, *(const v8h*)(pb + 16));
      acc = wmma16(am, bm, acc);
    }
    const int c = nt * 16 + col;
    const float bias = biasP[c];
    #pragma unroll
    for (int i = 0; i < 8; ++i) {
      int lr = mt * 16 + rbase + i;
      float v = acc[i] + bias;
      _Float16 hv = (lr < LM1 && c < CC) ? (_Float16)tanhf(v) : (_Float16)0.f;
      hout[(size_t)b * LP * CP + (size_t)lr * CP + c] = hv;
    }
  }
}

// -------------------------------------------------------------------
// K2: hAR = hA @ rand_matrix  ([304 x 128] x [128 x 128] per batch, WMMA).
// Padding regions are zero in both operands -> output padding is zero.
// -------------------------------------------------------------------
__global__ void __launch_bounds__(256) matmul_rand_kernel(
    const _Float16* __restrict__ hA, const _Float16* __restrict__ randT,
    _Float16* __restrict__ hR)
{
  const int b = blockIdx.x, tid = threadIdx.x;
  const int lane = tid & 31, wv = tid >> 5;
  const int col = lane & 15, half = lane >> 4;
  const int kb8 = half * 8, kb16 = half * 16, rbase = half * 8;
  const _Float16* A = hA + (size_t)b * LP * CP;
  _Float16* D = hR + (size_t)b * LP * CP;

  for (int t = wv; t < 19 * 8; t += 8) {
    const int mt = t >> 3, nt = t & 7;
    v8f acc = {};
    #pragma unroll
    for (int ks = 0; ks < 4; ++ks) {
      const _Float16* pa = A + (size_t)(mt * 16 + col) * CP + ks * 32 + kb8;
      v16h am = cat16(*(const v8h*)pa, *(const v8h*)(pa + 16));
      const _Float16* pb = randT + (size_t)(nt * 16 + col) * CP + ks * 32 + kb8;
      v16h bm = cat16(*(const v8h*)pb, *(const v8h*)(pb + 16));
      acc = wmma16(am, bm, acc);
    }
    #pragma unroll
    for (int i = 0; i < 8; ++i)
      D[(size_t)(mt * 16 + rbase + i) * CP + nt * 16 + col] = (_Float16)acc[i];
  }
}

// -------------------------------------------------------------------
// K3: fused attention.  S = tanh(hAR[b] @ hX[b]^T) [299x299], never
// materialized: row sums (att1 only) and column sums accumulate in LDS
// while D-tiles are live.  Operands staged in 158KB dynamic LDS.
// -------------------------------------------------------------------
__global__ void __launch_bounds__(256) attention_kernel(
    const _Float16* __restrict__ hR, const _Float16* __restrict__ hB,
    const _Float16* __restrict__ hN,
    float* __restrict__ rs1, float* __restrict__ cs1, float* __restrict__ cs3)
{
  extern __shared__ char smem[];
  _Float16* Ald = (_Float16*)smem;          // [304][128]
  _Float16* Bld = Ald + LP * CP;            // [304][128]
  float* rs = (float*)(smem + (size_t)2 * LP * CP * 2);
  float* cs = rs + LP;

  const int b = blockIdx.x, which = blockIdx.y, tid = threadIdx.x;
  const _Float16* gA = hR + (size_t)b * LP * CP;
  const _Float16* gB = (which == 0 ? hB : hN) + (size_t)b * LP * CP;

  {
    const uint4* sA = (const uint4*)gA;
    const uint4* sB = (const uint4*)gB;
    uint4* dA = (uint4*)Ald;
    uint4* dB = (uint4*)Bld;
    for (int i = tid; i < LP * CP / 8; i += 256) { dA[i] = sA[i]; dB[i] = sB[i]; }
  }
  for (int i = tid; i < LP; i += 256) { rs[i] = 0.f; cs[i] = 0.f; }
  __syncthreads();

  const int lane = tid & 31, wv = tid >> 5;
  const int col = lane & 15, half = (lane >> 4) & 1;
  const int kb8 = half * 8, kb16 = half * 16, rbase = half * 8;

  for (int t = wv; t < 19 * 19; t += 8) {
    const int mt = t / 19, nt = t % 19;
    v8f acc = {};
    #pragma unroll
    for (int ks = 0; ks < 4; ++ks) {
      const _Float16* pa = Ald + (mt * 16 + col) * CP + ks * 32 + kb8;
      v16h am = cat16(*(const v8h*)pa, *(const v8h*)(pa + 16));
      const _Float16* pb = Bld + (nt * 16 + col) * CP + ks * 32 + kb8;
      v16h bm = cat16(*(const v8h*)pb, *(const v8h*)(pb + 16));
      acc = wmma16(am, bm, acc);
    }
    float tv[8];
    float colp = 0.f;
    #pragma unroll
    for (int i = 0; i < 8; ++i) { tv[i] = tanhf(acc[i]); colp += tv[i]; }
    // column sums: lanes l and l+16 share column -> fold halves, one ds_add
    colp += __shfl_xor(colp, 16, 32);
    if (half == 0) atomicAdd(&cs[nt * 16 + col], colp);
    // row sums: reduce across the 16 lanes of each half, one ds_add per row
    #pragma unroll
    for (int i = 0; i < 8; ++i) {
      float r = tv[i];
      r += __shfl_xor(r, 1, 32);
      r += __shfl_xor(r, 2, 32);
      r += __shfl_xor(r, 4, 32);
      r += __shfl_xor(r, 8, 32);
      if (col == 0) atomicAdd(&rs[mt * 16 + rbase + i], r);
    }
  }
  __syncthreads();
  for (int i = tid; i < LP; i += 256) {
    if (which == 0) {
      rs1[(size_t)b * LP + i] = rs[i];
      cs1[(size_t)b * LP + i] = cs[i];
    } else {
      cs3[(size_t)b * LP + i] = cs[i];
    }
  }
}

// -------------------------------------------------------------------
// K4: softmax over means, weighted pooling, 8 log-sigmoid dots, loss.
// -------------------------------------------------------------------
__device__ __forceinline__ float logsig(float x) {
  return logf(1.f / (1.f + expf(-x)) + 0.001f);
}

__device__ float bred_max(float v, float* red) {
  int tid = threadIdx.x;
  red[tid] = v; __syncthreads();
  for (int s = 128; s > 0; s >>= 1) {
    if (tid < s) red[tid] = fmaxf(red[tid], red[tid + s]);
    __syncthreads();
  }
  float r = red[0]; __syncthreads();
  return r;
}

__device__ float bred_sum(float v, float* red) {
  int tid = threadIdx.x;
  red[tid] = v; __syncthreads();
  for (int s = 128; s > 0; s >>= 1) {
    if (tid < s) red[tid] += red[tid + s];
    __syncthreads();
  }
  float r = red[0]; __syncthreads();
  return r;
}

__device__ void softmax299(const float* __restrict__ src, float* w, float* red) {
  int tid = threadIdx.x;
  float lmax = -1e30f;
  for (int l = tid; l < LM1; l += 256) {
    float x = src[l] * (1.0f / (float)LM1);  // mean over 299
    w[l] = x;
    lmax = fmaxf(lmax, x);
  }
  float m = bred_max(lmax, red);
  float lsum = 0.f;
  for (int l = tid; l < LM1; l += 256) {
    float e = expf(w[l] - m);
    w[l] = e;
    lsum += e;
  }
  float s = bred_sum(lsum, red);
  float inv = 1.f / s;
  for (int l = tid; l < LM1; l += 256) w[l] *= inv;
  __syncthreads();
}

__global__ void __launch_bounds__(256) loss_kernel(
    const float* __restrict__ rs1, const float* __restrict__ cs1,
    const float* __restrict__ cs3,
    const _Float16* __restrict__ hA, const _Float16* __restrict__ hB,
    const _Float16* __restrict__ hN,
    const int* __restrict__ Na, const int* __restrict__ Nb, const int* __restrict__ Nn,
    const float* __restrict__ node_emb, float* __restrict__ out)
{
  __shared__ float wA[LP], wB[LP], wN[LP];
  __shared__ float red[256];
  __shared__ float cva[CC], cvb[CC], cvn[CC], na[CC], nb[CC], nn[CC];
  const int b = blockIdx.x, tid = threadIdx.x;

  softmax299(rs1 + (size_t)b * LP, wA, red);
  softmax299(cs1 + (size_t)b * LP, wB, red);
  softmax299(cs3 + (size_t)b * LP, wN, red);

  if (tid < CC) {
    const _Float16* pA = hA + (size_t)b * LP * CP + tid;
    const _Float16* pB = hB + (size_t)b * LP * CP + tid;
    const _Float16* pN = hN + (size_t)b * LP * CP + tid;
    float sa = 0.f, sb = 0.f, sn = 0.f;
    for (int l = 0; l < LM1; ++l) {
      sa += (float)pA[(size_t)l * CP] * wA[l];
      sb += (float)pB[(size_t)l * CP] * wB[l];
      sn += (float)pN[(size_t)l * CP] * wN[l];
    }
    cva[tid] = sa; cvb[tid] = sb; cvn[tid] = sn;
    na[tid] = node_emb[(size_t)Na[b] * EE + tid];
    nb[tid] = node_emb[(size_t)Nb[b] * EE + tid];
    nn[tid] = node_emb[(size_t)Nn[b] * EE + tid];
  }
  __syncthreads();

  if (tid == 0) {
    float d1 = 0, d2 = 0, d3 = 0, d4 = 0, d5 = 0, d6 = 0, d7 = 0, d8 = 0;
    for (int e = 0; e < CC; ++e) {
      d1 += cva[e] * cvb[e];  d2 += cva[e] * cvn[e];
      d3 += na[e] * nb[e];    d4 += na[e] * nn[e];
      d5 += cvb[e] * na[e];   d6 += cvn[e] * na[e];
      d7 += nb[e] * cva[e];   d8 += nb[e] * cvn[e];
    }
    float loss = RHO1 * (logsig(d1) + logsig(-d2))
               + RHO2 * (logsig(d3) + logsig(-d4))
               + RHO3 * (logsig(d5) + logsig(-d6))
               + RHO3 * (logsig(d7) + logsig(-d8));
    atomicAdd(out, -loss);
  }
}

// -------------------------------------------------------------------
extern "C" void kernel_launch(void* const* d_in, const int* in_sizes, int n_in,
                              void* d_out, int out_size, void* d_ws, size_t ws_size,
                              hipStream_t stream) {
  (void)in_sizes; (void)n_in; (void)out_size; (void)ws_size;
  const int*   Na       = (const int*)d_in[0];
  const int*   Nb       = (const int*)d_in[1];
  const int*   Nn       = (const int*)d_in[2];
  const int*   Ta       = (const int*)d_in[3];
  const int*   Tb       = (const int*)d_in[4];
  const int*   Tn       = (const int*)d_in[5];
  const float* text_emb = (const float*)d_in[6];
  const float* node_emb = (const float*)d_in[7];
  const float* conv_w   = (const float*)d_in[8];
  const float* conv_b   = (const float*)d_in[9];
  const float* rand_m   = (const float*)d_in[10];
  float* out = (float*)d_out;

  char* ws = (char*)d_ws;
  _Float16* Wp    = (_Float16*)(ws + WP_OFF);
  _Float16* randT = (_Float16*)(ws + RT_OFF);
  float*    biasP = (float*)(ws + BIAS_OFF);
  _Float16* hA    = (_Float16*)(ws + HA_OFF);
  _Float16* hB    = (_Float16*)(ws + HB_OFF);
  _Float16* hN    = (_Float16*)(ws + HN_OFF);
  _Float16* hR    = (_Float16*)(ws + HR_OFF);
  float* rs1 = (float*)(ws + RS1_OFF);
  float* cs1 = (float*)(ws + CS1_OFF);
  float* cs3 = (float*)(ws + CS3_OFF);

  // raise dynamic-LDS caps (CDNA5 WGP has 320KB LDS).  Host-side
  // attribute sets, not stream ops -- capture-safe.
  (void)hipFuncSetAttribute((const void*)attention_kernel,
                            hipFuncAttributeMaxDynamicSharedMemorySize,
                            (int)ATT_LDS);
  (void)hipFuncSetAttribute((const void*)conv_kernel,
                            hipFuncAttributeMaxDynamicSharedMemorySize,
                            (int)CONV_LDS);

  zero_out_kernel<<<1, 32, 0, stream>>>(out);
  prep_kernel<<<64, 256, 0, stream>>>(conv_w, rand_m, conv_b, Wp, randT, biasP);
  conv_kernel<<<dim3(BB, 3), 256, CONV_LDS, stream>>>(Ta, Tb, Tn, text_emb, biasP,
                                                      Wp, hA, hB, hN);
  matmul_rand_kernel<<<BB, 256, 0, stream>>>(hA, randT, hR);
  attention_kernel<<<dim3(BB, 2), 256, ATT_LDS, stream>>>(hR, hB, hN, rs1, cs1, cs3);
  loss_kernel<<<BB, 256, 0, stream>>>(rs1, cs1, cs3, hA, hB, hN,
                                      Na, Nb, Nn, node_emb, out);
}
